// EdgePoolBlock_5076651344272
// MI455X (gfx1250) — matrix operand order repaired and checked by reading, not compiled, not run on hardware
//
#include <hip/hip_runtime.h>
#include <stdint.h>

// ---------------- problem constants (match reference) ----------------
#define B_GRAPHS 64
#define N_NODES  1024
#define E_EDGES  1048576
#define D_DIM    128
#define EDGE_DIM 64
#define MSSG_DIM 32
#define GLOB_DIM 8
#define CNDT_DIM 8
#define ATTN_IN  112
#define K_PAD    128          // ATTN_IN padded to multiple of 32 for WMMA
#define HIDDEN   256
#define N_HEADS  4
#define HEAD_DIM 32
#define NUM_SEG  (B_GRAPHS * N_NODES)

typedef __attribute__((ext_vector_type(16))) __bf16 v16bf;
typedef __attribute__((ext_vector_type(8)))  float  v8f;

// ---------------- helpers ----------------
__device__ __forceinline__ unsigned short f32_to_bf16(float f) {
  unsigned u = __float_as_uint(f);
  u += 0x7fffu + ((u >> 16) & 1u);   // round-to-nearest-even
  return (unsigned short)(u >> 16);
}
__device__ __forceinline__ float bf16_to_f32(unsigned short h) {
  return __uint_as_float(((unsigned)h) << 16);
}
// order-preserving float<->uint map so atomicMax(uint) == float max
__device__ __forceinline__ unsigned enc_max(float f) {
  unsigned u = __float_as_uint(f);
  return (u & 0x80000000u) ? ~u : (u | 0x80000000u);
}
__device__ __forceinline__ float dec_max(unsigned k) {
  return (k & 0x80000000u) ? __uint_as_float(k & 0x7fffffffu)
                           : __uint_as_float(~k);
}

// ---------------- pass 0: init output / max-keys / denominators ----------------
__global__ void init_kernel(float* __restrict__ out, unsigned* __restrict__ mkey,
                            float* __restrict__ denom, int nout, int nseg4) {
  int i = blockIdx.x * blockDim.x + threadIdx.x;
  if (i < nout) out[i] = 0.f;
  if (i < nseg4) { mkey[i] = 0x007FFFFFu; /* == enc_max(-inf) */ denom[i] = 0.f; }
}

// ---------------- pass 1: W1 f32[112][256] -> W1T bf16[256][128] (transposed, K-padded)
__global__ void convert_w1_kernel(const float* __restrict__ W1,
                                  unsigned short* __restrict__ W1T) {
  int i = blockIdx.x * blockDim.x + threadIdx.x;   // over [n][k], 256*128
  if (i >= HIDDEN * K_PAD) return;
  int n = i >> 7, k = i & (K_PAD - 1);
  float v = (k < ATTN_IN) ? W1[k * HIDDEN + n] : 0.f;
  W1T[i] = f32_to_bf16(v);
}

// ---------------- pass 2: fused X-build + WMMA GEMM1 + GEMM2 + logits/segmax ----
// block = 256 threads = 8 waves; tile = 128 edges; grid = E/128
#define SM_X   0                            // bf16 [128][128]  32 KB
#define SM_H   32768                        // bf16 [128][256]  64 KB
#define SM_W1  98304                        // bf16 [256][128]  64 KB
#define SM_W2  163840                       // f32  [256*4]      4 KB
#define SM_B2  167936                       // f32  [4]
#define SM_B1  167952                       // f32  [256]        1 KB
#define SMEM_BYTES 168976

__global__ __launch_bounds__(256, 1)
void logits_kernel(const float* __restrict__ pooled_mssgs,   // [E][32]
                   const float* __restrict__ edges,          // [E][64]
                   const float* __restrict__ globs,          // [B][8]
                   const float* __restrict__ cndts,          // [B][8]
                   const unsigned short* __restrict__ W1T,   // [256][128] bf16
                   const float* __restrict__ b1,             // [256]
                   const float* __restrict__ W2,             // [256][4]
                   const float* __restrict__ b2,             // [4]
                   const int* __restrict__ batch_idx,        // [E]
                   const int* __restrict__ recv_idx,         // [E]
                   float* __restrict__ ws_logits,            // [E][4]
                   unsigned* __restrict__ mkey) {            // [NUM_SEG][4]
  extern __shared__ __align__(16) unsigned char smem[];
  unsigned short* Xs  = (unsigned short*)(smem + SM_X);
  unsigned short* Hs  = (unsigned short*)(smem + SM_H);
  unsigned short* W1s = (unsigned short*)(smem + SM_W1);
  float* W2s = (float*)(smem + SM_W2);
  float* b2s = (float*)(smem + SM_B2);
  float* b1s = (float*)(smem + SM_B1);

  const int t  = threadIdx.x;
  const int e0 = blockIdx.x * 128;

  // ---- stage W1T (64 KB) into LDS as 16B chunks ----
  {
    const uint4* src = (const uint4*)W1T;
    uint4* dst = (uint4*)W1s;
    #pragma unroll
    for (int i = t; i < (HIDDEN * K_PAD * 2) / 16; i += 256) dst[i] = src[i];
  }
  for (int i = t; i < HIDDEN * N_HEADS; i += 256) W2s[i] = W2[i];
  if (t < N_HEADS) b2s[t] = b2[t];
  if (t < HIDDEN)  b1s[t] = b1[t];

  // ---- build X tile [128 edges][128 K] bf16 (coalesced per source array) ----
  for (int i = t; i < 128 * MSSG_DIM; i += 256) {            // K 0..31
    int e = i >> 5, k = i & 31;
    Xs[e * K_PAD + k] = f32_to_bf16(pooled_mssgs[(size_t)e0 * MSSG_DIM + i]);
  }
  for (int i = t; i < 128 * EDGE_DIM; i += 256) {            // K 32..95
    int e = i >> 6, k = i & 63;
    Xs[e * K_PAD + MSSG_DIM + k] = f32_to_bf16(edges[(size_t)e0 * EDGE_DIM + i]);
  }
  for (int i = t; i < 128 * 32; i += 256) {                  // K 96..127 (ctxt + pad)
    int e = i >> 5, j = i & 31;
    float v = 0.f;
    if (j < GLOB_DIM + CNDT_DIM) {
      int b = batch_idx[e0 + e];
      v = (j < GLOB_DIM) ? globs[b * GLOB_DIM + j]
                         : cndts[b * CNDT_DIM + (j - GLOB_DIM)];
    }
    Xs[e * K_PAD + (MSSG_DIM + EDGE_DIM) + j] = f32_to_bf16(v);
  }
  __syncthreads();

  // ---- GEMM1: each wave owns a 16-edge M-tile, all 16 N-tiles, K=128 ----
  const int wave  = t >> 5;
  const int lane  = t & 31;
  const int mrow  = lane & 15;
  const int ahalf = lane >> 4;                 // A: K-split 0/8; B: K-split 0/8
  const unsigned short* xrow = Xs + (wave * 16 + mrow) * K_PAD;

  v8f acc[16];
  #pragma unroll
  for (int nt = 0; nt < 16; ++nt)
    #pragma unroll
    for (int v = 0; v < 8; ++v) acc[nt][v] = 0.f;

  #pragma unroll
  for (int kk = 0; kk < K_PAD / 32; ++kk) {
    // A fragment: lane holds row m, K = kb..kb+7 and kb+16..kb+23 (kb = kk*32 + 16-lane-half*8)
    union { uint4 q[2]; v16bf v; } A;
    int kb = kk * 32 + ahalf * 8;
    A.q[0] = *(const uint4*)(xrow + kb);
    A.q[1] = *(const uint4*)(xrow + kb + 16);
    #pragma unroll
    for (int nt = 0; nt < 16; ++nt) {
      // B fragment: lane holds col n = nt*16 + (lane&15), K = kb..kb+7 and kb+16..kb+23 (kb = kk*32 + half*8)
      union { uint4 q[2]; v16bf v; } Bf;
      const unsigned short* wrow =
          W1s + (nt * 16 + mrow) * K_PAD + kk * 32 + ahalf * 8;
      Bf.q[0] = *(const uint4*)(wrow);
      Bf.q[1] = *(const uint4*)(wrow + 16);
      acc[nt] = __builtin_amdgcn_wmma_f32_16x16x32_bf16(
          false, A.v, false, Bf.v, (short)0, acc[nt], false, false);
    }
  }

  // ---- bias + leaky_relu(0.1), write h tile to LDS as bf16 ----
  #pragma unroll
  for (int nt = 0; nt < 16; ++nt) {
    int n = nt * 16 + mrow;                    // C/D layout: col = lane&15
    float bias = b1s[n];
    #pragma unroll
    for (int v = 0; v < 8; ++v) {
      float h = acc[nt][v] + bias;
      h = (h >= 0.f) ? h : 0.1f * h;
      int m = v + 8 * ahalf;                   // C/D layout: row = vgpr + 8*(lane>=16)
      Hs[(wave * 16 + m) * HIDDEN + n] = f32_to_bf16(h);
    }
  }
  __syncthreads();

  // ---- GEMM2 (N_HEADS=4): tiny, do as scalar dots from LDS; logits + seg-max ----
  for (int o = t; o < 128 * N_HEADS; o += 256) {
    int e = o >> 2, hIdx = o & 3;
    float s = b2s[hIdx];
    const unsigned short* hrow = Hs + e * HIDDEN;
    #pragma unroll 8
    for (int n = 0; n < HIDDEN; ++n)
      s += bf16_to_f32(hrow[n]) * W2s[n * N_HEADS + hIdx];
    int ge  = e0 + e;
    int seg = batch_idx[ge] * N_NODES + recv_idx[ge];
    ws_logits[(size_t)ge * N_HEADS + hIdx] = s;
    atomicMax(&mkey[seg * N_HEADS + hIdx], enc_max(s));
  }
}

// ---------------- pass 3: ex = exp(logit - segmax); denom += ex ----------------
__global__ void exps_kernel(const int* __restrict__ batch_idx,
                            const int* __restrict__ recv_idx,
                            float* __restrict__ ws_logits,        // in logits, out ex
                            const unsigned* __restrict__ mkey,
                            float* __restrict__ denom) {
  int i = blockIdx.x * blockDim.x + threadIdx.x;
  if (i >= E_EDGES * N_HEADS) return;
  int ge = i >> 2, hIdx = i & 3;
  int seg = batch_idx[ge] * N_NODES + recv_idx[ge];
  float m  = dec_max(mkey[seg * N_HEADS + hIdx]);
  float ex = __expf(ws_logits[i] - m);
  ws_logits[i] = ex;
  atomicAdd(&denom[seg * N_HEADS + hIdx], ex);
}

// ---------------- pass 4: out[seg][d] += new_edges[e][d] * attn[head(d)] -------
// one wave per edge; lane covers 4 consecutive d (head = lane>>3 since HEAD_DIM=32)
__global__ __launch_bounds__(256)
void scatter_kernel(const float* __restrict__ new_edges,   // [E][128]
                    const float* __restrict__ ws_ex,       // [E][4]
                    const float* __restrict__ denom,       // [NUM_SEG][4]
                    const int* __restrict__ batch_idx,
                    const int* __restrict__ recv_idx,
                    float* __restrict__ out) {             // [B*N][128]
  int wave = threadIdx.x >> 5;
  int lane = threadIdx.x & 31;
  int ge   = blockIdx.x * 8 + wave;
  if (ge >= E_EDGES) return;
  int seg  = batch_idx[ge] * N_NODES + recv_idx[ge];
  int hIdx = lane >> 3;
  float attn = ws_ex[(size_t)ge * N_HEADS + hIdx] /
               denom[seg * N_HEADS + hIdx];
  const float* src = new_edges + (size_t)ge * D_DIM + lane * 4;
  __builtin_prefetch(src + 8 * D_DIM, 0, 0);               // global_prefetch_b8
  float4 ne = *(const float4*)src;
  float* op = out + (size_t)seg * D_DIM + lane * 4;
  atomicAdd(op + 0, ne.x * attn);
  atomicAdd(op + 1, ne.y * attn);
  atomicAdd(op + 2, ne.z * attn);
  atomicAdd(op + 3, ne.w * attn);
}

// ---------------- launcher ----------------
extern "C" void kernel_launch(void* const* d_in, const int* in_sizes, int n_in,
                              void* d_out, int out_size, void* d_ws, size_t ws_size,
                              hipStream_t stream) {
  const float* new_edges    = (const float*)d_in[0];
  const float* edges        = (const float*)d_in[1];
  const float* pooled_mssgs = (const float*)d_in[2];
  const float* globs        = (const float*)d_in[3];
  const float* cndts        = (const float*)d_in[4];
  const float* W1           = (const float*)d_in[5];
  const float* b1           = (const float*)d_in[6];
  const float* W2           = (const float*)d_in[7];
  const float* b2           = (const float*)d_in[8];
  const int*   batch_idx    = (const int*)d_in[9];
  const int*   recv_idx     = (const int*)d_in[10];
  float* out = (float*)d_out;

  // workspace carve-up
  unsigned char* ws = (unsigned char*)d_ws;
  unsigned short* W1T   = (unsigned short*)(ws);                     // 64 KB
  float*    ws_logits   = (float*)(ws + 65536);                      // 16 MB
  unsigned* mkey        = (unsigned*)(ws + 65536 + (size_t)E_EDGES * N_HEADS * 4);
  float*    denom       = (float*)((unsigned char*)mkey + (size_t)NUM_SEG * N_HEADS * 4);

  const int nout  = B_GRAPHS * N_NODES * D_DIM;       // 8388608
  const int nseg4 = NUM_SEG * N_HEADS;                // 262144

  init_kernel<<<(nout + 255) / 256, 256, 0, stream>>>(out, mkey, denom, nout, nseg4);
  convert_w1_kernel<<<(HIDDEN * K_PAD + 255) / 256, 256, 0, stream>>>(W1, W1T);
  logits_kernel<<<E_EDGES / 128, 256, SMEM_BYTES, stream>>>(
      pooled_mssgs, edges, globs, cndts, W1T, b1, W2, b2,
      batch_idx, recv_idx, ws_logits, mkey);
  exps_kernel<<<(E_EDGES * N_HEADS + 255) / 256, 256, 0, stream>>>(
      batch_idx, recv_idx, ws_logits, mkey, denom);
  scatter_kernel<<<E_EDGES / 8, 256, 0, stream>>>(
      new_edges, ws_logits, denom, batch_idx, recv_idx, out);
}
